// combined_model_11398843203893
// MI455X (gfx1250) — matrix operand order repaired and checked by reading, not compiled, not run on hardware
//
#include <hip/hip_runtime.h>

// ---------------------------------------------------------------------------
// Types for CDNA5 WMMA (wave32, v_wmma_f32_16x16x32_bf16)
// ---------------------------------------------------------------------------
typedef __bf16 bf16;
typedef __attribute__((ext_vector_type(16))) __bf16 v16bf;
typedef __attribute__((ext_vector_type(8)))  __bf16 v8bf;
typedef __attribute__((ext_vector_type(8)))  float  v8f;

__device__ __forceinline__ bf16 f2bf(float f) {
  unsigned u = __float_as_uint(f);
  u += 0x7FFFu + ((u >> 16) & 1u);           // round-to-nearest-even
  unsigned short s = (unsigned short)(u >> 16);
  return __builtin_bit_cast(bf16, s);
}

__device__ __forceinline__ int f2ord(float f) {
  int i = __float_as_int(f);
  return i >= 0 ? i : (i ^ 0x7FFFFFFF);
}
__device__ __forceinline__ float ord2f(int i) {
  return __int_as_float(i >= 0 ? i : (i ^ 0x7FFFFFFF));
}

// ---------------------------------------------------------------------------
// Small utility kernels
// ---------------------------------------------------------------------------
__global__ void fill_f32_kernel(float* p, float v, int n) {
  int i = blockIdx.x * blockDim.x + threadIdx.x;
  if (i < n) p[i] = v;
}
__global__ void fill_i32_kernel(int* p, int v, int n) {
  int i = blockIdx.x * blockDim.x + threadIdx.x;
  if (i < n) p[i] = v;
}
__global__ void convert_bf16_kernel(const float* __restrict__ src,
                                    bf16* __restrict__ dst, int n, int do_relu) {
  int i = blockIdx.x * blockDim.x + threadIdx.x;
  if (i >= n) return;
  float v = src[i];
  if (do_relu) v = v > 0.f ? v : 0.f;
  dst[i] = f2bf(v);
}

// Transpose+pad fp32 weight [k_rows x n_cols] (row-major, ld_src) into
// column-major bf16 Wt[n][K]: Wt[n*ld_dst + k_off + kk], zero-padded.
__global__ void prep_wt_kernel(const float* __restrict__ W, int ld_src,
                               int k_rows, int n_cols,
                               bf16* __restrict__ Wt, int ld_dst,
                               int k_off, int k_zone, int n_pad) {
  int idx = blockIdx.x * blockDim.x + threadIdx.x;
  int total = n_pad * k_zone;
  if (idx >= total) return;
  int n  = idx / k_zone;
  int kk = idx - n * k_zone;
  float v = (kk < k_rows && n < n_cols) ? W[(size_t)kk * ld_src + n] : 0.0f;
  Wt[(size_t)n * ld_dst + k_off + kk] = f2bf(v);
}

// ---------------------------------------------------------------------------
// WMMA GEMM: C[M,256] = A[M,Kpad](bf16,row-major) @ W(col-major bf16 Wt[256][Kpad]) + bias
// block = 128 threads (4 waves); each wave -> 16 rows x 64 cols (4 accum tiles)
// grid = (ceil(M/64), 256/64)
// ---------------------------------------------------------------------------
__global__ void gemm_bias_wmma_kernel(const bf16* __restrict__ A, int lda,
                                      const bf16* __restrict__ Wt, int ldb,
                                      const float* __restrict__ bias, int nbias,
                                      float* __restrict__ C, int ldc,
                                      int M, int Kpad) {
  const int lane  = threadIdx.x & 31;
  const int wave  = threadIdx.x >> 5;              // 0..3
  const int mtile = blockIdx.x * 64 + wave * 16;
  const int ncol0 = blockIdx.y * 64;
  const int nlane = lane & 15;
  const int khalf = lane >> 4;

  int mrow = mtile + nlane;
  if (mrow >= M) mrow = M - 1;                     // clamp (keeps EXEC full)
  const bf16* arow = A + (size_t)mrow * lda;

  v8f acc[4] = {};
  for (int kb = 0; kb < Kpad; kb += 32) {
    union { v16bf v; v8bf h[2]; } ua;
    ua.h[0] = *(const v8bf*)(arow + kb + khalf * 16);
    ua.h[1] = *(const v8bf*)(arow + kb + khalf * 16 + 8);
    const bf16* wb = Wt + kb + khalf * 16;
#pragma unroll
    for (int f = 0; f < 4; ++f) {
      v16bf b = *(const v16bf*)(wb + (size_t)(ncol0 + f * 16 + nlane) * ldb);
      acc[f] = __builtin_amdgcn_wmma_f32_16x16x32_bf16(
          false, ua.v, false, b, (short)0, acc[f], false, false);
    }
  }

  const int mbase = mtile + (khalf ? 8 : 0);
#pragma unroll
  for (int f = 0; f < 4; ++f) {
    int n = ncol0 + f * 16 + nlane;
    float bv = (n < nbias) ? bias[n] : 0.0f;
#pragma unroll
    for (int r = 0; r < 8; ++r) {
      int m = mbase + r;
      if (m < M) C[(size_t)m * ldc + n] = acc[f][r] + bv;
    }
  }
}

// ---------------------------------------------------------------------------
// Edge kernels (E2 = E + N self loops). xl/xr/agg have row stride 256.
// ---------------------------------------------------------------------------
__global__ void edge_logits_kernel(const float* __restrict__ xl,
                                   const float* __restrict__ xr,
                                   const int* __restrict__ ei, int E, int N,
                                   const float* __restrict__ att, int D,
                                   float* __restrict__ elog,
                                   int* __restrict__ mmax) {
  int lane = threadIdx.x & 31;
  int e = blockIdx.x * (blockDim.x >> 5) + (threadIdx.x >> 5);
  int E2 = E + N;
  if (e >= E2) return;
  int s, d;
  if (e < E) { s = ei[e]; d = ei[E + e]; } else { s = d = e - E; }
  const float* pl = xl + (size_t)s * 256;
  const float* pr = xr + (size_t)d * 256;
  float dot = 0.f;
  for (int k = lane; k < D; k += 32) {
    float v = pl[k] + pr[k];
    v = v > 0.f ? v : 0.2f * v;                    // leaky_relu
    dot += att[k] * v;
  }
#pragma unroll
  for (int m = 16; m >= 1; m >>= 1) dot += __shfl_xor(dot, m, 32);
  if (lane == 0) {
    elog[e] = dot;
    atomicMax(mmax + d, f2ord(dot));
  }
}

__global__ void edge_alpha_kernel(const float* __restrict__ elog,
                                  const int* __restrict__ ei, int E, int N,
                                  const int* __restrict__ mmax,
                                  float* __restrict__ ssum,
                                  float* __restrict__ alpha) {
  int e = blockIdx.x * blockDim.x + threadIdx.x;
  int E2 = E + N;
  if (e >= E2) return;
  int d = (e < E) ? ei[E + e] : (e - E);
  float a = __expf(elog[e] - ord2f(mmax[d]));
  alpha[e] = a;
  atomicAdd(ssum + d, a);
}

__global__ void edge_scatter_kernel(const float* __restrict__ xl,
                                    const int* __restrict__ ei, int E, int N,
                                    const float* __restrict__ alpha,
                                    const float* __restrict__ ssum, int D,
                                    float* __restrict__ agg) {
  int lane = threadIdx.x & 31;
  int e = blockIdx.x * (blockDim.x >> 5) + (threadIdx.x >> 5);
  int E2 = E + N;
  if (e >= E2) return;
  int s, d;
  if (e < E) { s = ei[e]; d = ei[E + e]; } else { s = d = e - E; }
  float w = alpha[e] / ssum[d];
  const float* pl = xl + (size_t)s * 256;
  float* po = agg + (size_t)d * 256;
  for (int k = lane; k < D; k += 32) atomicAdd(po + k, w * pl[k]);
}

// ---------------------------------------------------------------------------
// Fused head: out[q] = relu( relu(concat(emb[q0],emb[q1]) @ Wm1 + bm1) @ Wm2 + bm2 )
// emb bf16 [N,256] zero-padded; Wm1 pre-transposed+padded -> wm1t[256][512].
// block = 256 threads (8 waves); each wave -> 16 query rows.
// ---------------------------------------------------------------------------
__global__ void head_kernel(const bf16* __restrict__ embbf,
                            const int* __restrict__ query, int Q,
                            const bf16* __restrict__ wm1t,
                            const float* __restrict__ bm1,
                            const float* __restrict__ wm2,
                            const float* __restrict__ bm2,
                            float* __restrict__ out) {
  const int lane  = threadIdx.x & 31;
  const int wave  = threadIdx.x >> 5;              // 0..7
  const int qbase = (blockIdx.x * 8 + wave) * 16;
  const int nlane = lane & 15;
  const int khalf = lane >> 4;

  int qm = qbase + nlane;
  if (qm >= Q) qm = Q - 1;                         // clamp (EXEC stays full)
  const int q0 = query[2 * qm];
  const int q1 = query[2 * qm + 1];
  const bf16* row0 = embbf + (size_t)q0 * 256;
  const bf16* row1 = embbf + (size_t)q1 * 256;

  v8f partial = {};
  for (int nb = 0; nb < 4; ++nb) {
    const int ncol0 = nb * 64;
    v8f acc[4] = {};
    for (int kb = 0; kb < 512; kb += 32) {
      const bf16* base = (kb < 256) ? (row0 + kb) : (row1 + (kb - 256));
      union { v16bf v; v8bf h[2]; } ua;
      ua.h[0] = *(const v8bf*)(base + khalf * 16);
      ua.h[1] = *(const v8bf*)(base + khalf * 16 + 8);
      const bf16* wb = wm1t + kb + khalf * 16;
#pragma unroll
      for (int f = 0; f < 4; ++f) {
        v16bf b = *(const v16bf*)(wb + (size_t)(ncol0 + f * 16 + nlane) * 512);
        acc[f] = __builtin_amdgcn_wmma_f32_16x16x32_bf16(
            false, ua.v, false, b, (short)0, acc[f], false, false);
      }
    }
#pragma unroll
    for (int f = 0; f < 4; ++f) {
      int n = ncol0 + f * 16 + nlane;
      float bv = bm1[n];
      float wv = wm2[n];
#pragma unroll
      for (int r = 0; r < 8; ++r) {
        float h = acc[f][r] + bv;
        h = h > 0.f ? h : 0.f;                     // relu(hidden)
        partial[r] += h * wv;                      // fold in @Wm2
      }
    }
  }

  const float b2 = bm2[0];
#pragma unroll
  for (int r = 0; r < 8; ++r) {
    float v = partial[r];
    v += __shfl_xor(v, 1, 32);
    v += __shfl_xor(v, 2, 32);
    v += __shfl_xor(v, 4, 32);
    v += __shfl_xor(v, 8, 32);                     // sum over the 16 N-lanes
    if (nlane == 0) {
      int m = qbase + r + (khalf ? 8 : 0);
      if (m < Q) {
        float o = v + b2;
        out[m] = o > 0.f ? o : 0.f;
      }
    }
  }
}

// ---------------------------------------------------------------------------
// Host orchestration
// ---------------------------------------------------------------------------
extern "C" void kernel_launch(void* const* d_in, const int* in_sizes, int n_in,
                              void* d_out, int out_size, void* d_ws, size_t ws_size,
                              hipStream_t stream) {
  const float* x    = (const float*)d_in[0];
  const int*   ei   = (const int*)d_in[1];
  const int*   qry  = (const int*)d_in[2];
  const float* W1l  = (const float*)d_in[3];
  const float* b1l  = (const float*)d_in[4];
  const float* W1r  = (const float*)d_in[5];
  const float* b1r  = (const float*)d_in[6];
  const float* att1 = (const float*)d_in[7];
  const float* W2l  = (const float*)d_in[8];
  const float* b2l  = (const float*)d_in[9];
  const float* W2r  = (const float*)d_in[10];
  const float* b2r  = (const float*)d_in[11];
  const float* att2 = (const float*)d_in[12];
  const float* Wm1  = (const float*)d_in[13];
  const float* bm1  = (const float*)d_in[14];
  const float* Wm2  = (const float*)d_in[15];
  const float* bm2  = (const float*)d_in[16];
  float* out = (float*)d_out;

  const int F  = 128;
  const int N  = in_sizes[0] / F;
  const int E  = in_sizes[1] / 2;
  const int Q  = in_sizes[2] / 2;
  const int D  = in_sizes[4];          // 250
  const int E2 = E + N;

  // workspace carve-out
  char* ws = (char*)d_ws;
  size_t off = 0;
  auto carve = [&](size_t bytes) -> char* {
    char* p = ws + off;
    off += (bytes + 255) & ~(size_t)255;
    return p;
  };
  bf16*  xbf   = (bf16*)carve((size_t)N * F * 2);
  bf16*  w1lt  = (bf16*)carve(256 * 128 * 2);
  bf16*  w1rt  = (bf16*)carve(256 * 128 * 2);
  bf16*  w2lt  = (bf16*)carve(256 * 256 * 2);
  bf16*  w2rt  = (bf16*)carve(256 * 256 * 2);
  bf16*  wm1t  = (bf16*)carve(256 * 512 * 2);
  float* xl    = (float*)carve((size_t)N * 256 * 4);
  float* xr    = (float*)carve((size_t)N * 256 * 4);
  float* agg   = (float*)carve((size_t)N * 256 * 4);
  bf16*  hbf   = (bf16*)carve((size_t)N * 256 * 2);
  bf16*  embbf = (bf16*)carve((size_t)N * 256 * 2);
  float* elog  = (float*)carve((size_t)E2 * 4);
  float* alpha = (float*)carve((size_t)E2 * 4);
  int*   mmax  = (int*)carve((size_t)N * 4);
  float* ssum  = (float*)carve((size_t)N * 4);

  const int NEG_INF_ORD = (int)0x807FFFFFu;        // f2ord(-inf)
  auto cdiv = [](int a, int b) { return (a + b - 1) / b; };

  // --- weight prep (bf16, transposed, padded) ---
  convert_bf16_kernel<<<cdiv(N * F, 256), 256, 0, stream>>>(x, xbf, N * F, 0);
  prep_wt_kernel<<<cdiv(256 * 128, 256), 256, 0, stream>>>(W1l, D, F, D, w1lt, 128, 0, 128, 256);
  prep_wt_kernel<<<cdiv(256 * 128, 256), 256, 0, stream>>>(W1r, D, F, D, w1rt, 128, 0, 128, 256);
  prep_wt_kernel<<<cdiv(256 * 256, 256), 256, 0, stream>>>(W2l, D, D, D, w2lt, 256, 0, 256, 256);
  prep_wt_kernel<<<cdiv(256 * 256, 256), 256, 0, stream>>>(W2r, D, D, D, w2rt, 256, 0, 256, 256);
  prep_wt_kernel<<<cdiv(256 * 256, 256), 256, 0, stream>>>(Wm1,           256, D, 256, wm1t, 512, 0,   256, 256);
  prep_wt_kernel<<<cdiv(256 * 256, 256), 256, 0, stream>>>(Wm1 + (size_t)D * 256, 256, D, 256, wm1t, 512, 256, 256, 256);

  dim3 ggrid(cdiv(N, 64), 4);

  // --- layer 1 ---
  gemm_bias_wmma_kernel<<<ggrid, 128, 0, stream>>>(xbf, 128, w1lt, 128, b1l, D, xl, 256, N, 128);
  gemm_bias_wmma_kernel<<<ggrid, 128, 0, stream>>>(xbf, 128, w1rt, 128, b1r, D, xr, 256, N, 128);
  fill_i32_kernel<<<cdiv(N, 256), 256, 0, stream>>>(mmax, NEG_INF_ORD, N);
  fill_f32_kernel<<<cdiv(N, 256), 256, 0, stream>>>(ssum, 0.f, N);
  fill_f32_kernel<<<cdiv(N * 256, 256), 256, 0, stream>>>(agg, 0.f, N * 256);
  edge_logits_kernel<<<cdiv(E2, 8), 256, 0, stream>>>(xl, xr, ei, E, N, att1, D, elog, mmax);
  edge_alpha_kernel<<<cdiv(E2, 256), 256, 0, stream>>>(elog, ei, E, N, mmax, ssum, alpha);
  edge_scatter_kernel<<<cdiv(E2, 8), 256, 0, stream>>>(xl, ei, E, N, alpha, ssum, D, agg);
  convert_bf16_kernel<<<cdiv(N * 256, 256), 256, 0, stream>>>(agg, hbf, N * 256, 1); // relu

  // --- layer 2 ---
  gemm_bias_wmma_kernel<<<ggrid, 128, 0, stream>>>(hbf, 256, w2lt, 256, b2l, D, xl, 256, N, 256);
  gemm_bias_wmma_kernel<<<ggrid, 128, 0, stream>>>(hbf, 256, w2rt, 256, b2r, D, xr, 256, N, 256);
  fill_i32_kernel<<<cdiv(N, 256), 256, 0, stream>>>(mmax, NEG_INF_ORD, N);
  fill_f32_kernel<<<cdiv(N, 256), 256, 0, stream>>>(ssum, 0.f, N);
  fill_f32_kernel<<<cdiv(N * 256, 256), 256, 0, stream>>>(agg, 0.f, N * 256);
  edge_logits_kernel<<<cdiv(E2, 8), 256, 0, stream>>>(xl, xr, ei, E, N, att2, D, elog, mmax);
  edge_alpha_kernel<<<cdiv(E2, 256), 256, 0, stream>>>(elog, ei, E, N, mmax, ssum, alpha);
  edge_scatter_kernel<<<cdiv(E2, 8), 256, 0, stream>>>(xl, ei, E, N, alpha, ssum, D, agg);
  convert_bf16_kernel<<<cdiv(N * 256, 256), 256, 0, stream>>>(agg, embbf, N * 256, 0);

  // --- fused head ---
  head_kernel<<<cdiv(Q, 128), 256, 0, stream>>>(embbf, qry, Q, wm1t, bm1, Wm2, bm2, out);
}
